// SoftAssigner_17257178595386
// MI455X (gfx1250) — matrix operand order repaired and checked by reading, not compiled, not run on hardware
//
#include <hip/hip_runtime.h>
#include <hip/hip_bf16.h>

// ---------------------------------------------------------------------------
// Problem dims (fixed by the reference)
// ---------------------------------------------------------------------------
#define M_CTX   1024
#define N_SMP   16384
#define D_IN    1024
#define D_MODEL 1024

typedef unsigned short u16;
typedef __attribute__((ext_vector_type(16))) __bf16        v16bf;
typedef __attribute__((ext_vector_type(8)))  float         v8f;
typedef __attribute__((ext_vector_type(4)))  unsigned int  u32x4;
typedef __attribute__((ext_vector_type(4)))  int           v4i;

union FragAB { v16bf v; u32x4 u[2]; };   // 32 bytes = 8 VGPRs, wave32 bf16 A/B frag

__device__ __forceinline__ u16 f2bf(float x) {
    unsigned u = __float_as_uint(x);
    u += 0x7fffu + ((u >> 16) & 1u);          // round-to-nearest-even
    return (u16)(u >> 16);
}

// ---------------------------------------------------------------------------
// CDNA5 async global->LDS 16B copy (GLOBAL_LOAD_ASYNC_TO_LDS_B128, ASYNCcnt)
// builtin signature: (v4i addrspace(1)*, v4i addrspace(3)*, imm offset, imm cpol)
// ---------------------------------------------------------------------------
__device__ __forceinline__ void async_copy16(const u16* g, u16* l) {
#if __has_builtin(__builtin_amdgcn_global_load_async_to_lds_b128)
    __builtin_amdgcn_global_load_async_to_lds_b128(
        (__attribute__((address_space(1))) v4i*)(void*)g,
        (__attribute__((address_space(3))) v4i*)(void*)l, 0, 0);
#else
    unsigned lofs = (unsigned)(size_t)(__attribute__((address_space(3))) void*)(void*)l;
    asm volatile("global_load_async_to_lds_b128 %0, %1, off"
                 :: "v"(lofs), "v"(g) : "memory");
#endif
}

__device__ __forceinline__ void wait_async0() {
#if __has_builtin(__builtin_amdgcn_s_wait_asynccnt)
    __builtin_amdgcn_s_wait_asynccnt(0);
#else
    asm volatile("s_wait_asynccnt 0" ::: "memory");
#endif
}

// ---------------------------------------------------------------------------
// Elementwise casts
// ---------------------------------------------------------------------------
__global__ void cast_bf16_kernel(const float* __restrict__ in, u16* __restrict__ out, int n) {
    int i = blockIdx.x * 256 + threadIdx.x;
    if (i < n) out[i] = f2bf(in[i]);
}

// out[c*R + r] = bf16(in[r*C + c])  (writes coalesced)
__global__ void transpose_cast_kernel(const float* __restrict__ in, u16* __restrict__ out,
                                      int R, int C) {
    long long o = (long long)blockIdx.x * 256 + threadIdx.x;
    if (o >= (long long)R * C) return;
    int c = (int)(o / R);
    int r = (int)(o % R);
    out[o] = f2bf(in[(long long)r * C + c]);
}

// ---------------------------------------------------------------------------
// Tiled bf16 WMMA GEMM:  C[i][j] = sum_k A[i][k] * B[j][k]  (+ bias[j])
//   A: M x K bf16 row-major (lda), B: N x K bf16 row-major (ldb)
//   outF: f32 row-major (ldF) if non-null
//   outH: bf16, normal (i*ldH+j) or transposed (j*ldH+i) if non-null
// Block tile 128x128, BK=32; 8 waves (256 thr) in 4(M)x2(N); wave tile 32x64
// via 2x4 WMMA accumulators. Staging via async global->LDS B128 (4/thread).
// M % 128 == 0, N % 128 == 0, K % 32 == 0 (all shapes here satisfy this).
// ---------------------------------------------------------------------------
#define BM 128
#define BN 128
#define BK 32
#define LSTR 40   // LDS row stride in halves: 80B -> 16B aligned, spreads banks

__global__ __launch_bounds__(256)
void gemm_bt_kernel(const u16* __restrict__ A, int lda,
                    const u16* __restrict__ B, int ldb,
                    const float* __restrict__ bias,
                    float* __restrict__ outF, int ldF,
                    u16* __restrict__ outH, int ldH, int transH,
                    int K) {
    __shared__ u16 lA[BM * LSTR];
    __shared__ u16 lB[BN * LSTR];

    const int t    = threadIdx.x;
    const int lane = t & 31;
    const int w    = t >> 5;
    const int wm   = w & 3;    // 4 waves along M
    const int wn   = w >> 2;   // 2 waves along N (64 cols each)
    const int bm   = blockIdx.y * BM;
    const int bn   = blockIdx.x * BN;

    // staging coordinates (4 B128 async copies per thread, no branches)
    const int r0 = t >> 2;            // 0..63
    const int c0 = (t & 3) * 8;       // halves
    const int r1 = r0 + 64;

    v8f c[2][4] = {};

    for (int k0 = 0; k0 < K; k0 += BK) {
        // --- async stage A tile (128x32 halves) and B tile (128x32 halves) ---
        async_copy16(&A[(size_t)(bm + r0) * lda + k0 + c0], &lA[r0 * LSTR + c0]);
        async_copy16(&A[(size_t)(bm + r1) * lda + k0 + c0], &lA[r1 * LSTR + c0]);
        async_copy16(&B[(size_t)(bn + r0) * ldb + k0 + c0], &lB[r0 * LSTR + c0]);
        async_copy16(&B[(size_t)(bn + r1) * ldb + k0 + c0], &lB[r1 * LSTR + c0]);
        wait_async0();
        __syncthreads();

        // --- fragment loads per ISA 7.12.2 bf16 layouts ---
        // A (16x32): lane holds row (lane&15), K = {kb..kb+7, kb+16..kb+23}, kb=(lane>>4)*8
        FragAB fa[2], fb[4];
        const int ar = wm * 32 + (lane & 15);
        const int kb = (lane >> 4) * 8;
        fa[0].u[0] = *(const u32x4*)&lA[(ar)      * LSTR + kb];
        fa[0].u[1] = *(const u32x4*)&lA[(ar)      * LSTR + kb + 16];
        fa[1].u[0] = *(const u32x4*)&lA[(ar + 16) * LSTR + kb];
        fa[1].u[1] = *(const u32x4*)&lA[(ar + 16) * LSTR + kb + 16];
        // B (32x16): lane holds col (lane&15), K = {kb2..kb2+7, kb2+16..kb2+23}, kb2=(lane>>4)*8
        const int kb2 = (lane >> 4) * 8;
#pragma unroll
        for (int j = 0; j < 4; ++j) {
            const int bc = wn * 64 + j * 16 + (lane & 15);
            fb[j].u[0] = *(const u32x4*)&lB[bc * LSTR + kb2];
            fb[j].u[1] = *(const u32x4*)&lB[bc * LSTR + kb2 + 16];
        }

#pragma unroll
        for (int i = 0; i < 2; ++i)
#pragma unroll
            for (int j = 0; j < 4; ++j)
                c[i][j] = __builtin_amdgcn_wmma_f32_16x16x32_bf16(
                    false, fa[i].v, false, fb[j].v, (short)0, c[i][j], false, false);
        __syncthreads();
    }

    // --- epilogue: C/D layout — VGPR v: M = v + (lane>>4)*8, N = lane&15 ---
#pragma unroll
    for (int i = 0; i < 2; ++i)
#pragma unroll
        for (int j = 0; j < 4; ++j)
#pragma unroll
            for (int vv = 0; vv < 8; ++vv) {
                int gi = bm + wm * 32 + i * 16 + (lane >> 4) * 8 + vv;
                int gj = bn + wn * 64 + j * 16 + (lane & 15);
                float val = c[i][j][vv];
                if (bias) val += bias[gj];
                if (outF) outF[(size_t)gi * ldF + gj] = val;
                if (outH) {
                    if (transH) outH[(size_t)gj * ldH + gi] = f2bf(val);
                    else        outH[(size_t)gi * ldH + gj] = f2bf(val);
                }
            }
}

// ---------------------------------------------------------------------------
// Gumbel-softmax per sample row n (block = row, 256 thr, M_CTX=1024 -> 4/thr)
//   soft  (u1): y = softmax(2*(logit+g1)) -> asgT[m][n] bf16 (transposed)
//   hard  (u2): one-hot(argmax(logit+g2)) -> hard[n][m] f32
// ---------------------------------------------------------------------------
__global__ __launch_bounds__(256)
void gumbel_kernel(const float* __restrict__ logits,
                   const float* __restrict__ u1, const float* __restrict__ u2,
                   u16* __restrict__ asgT, float* __restrict__ hard) {
    const int n = blockIdx.x;
    const int t = threadIdx.x;
    __shared__ float sv[256];
    __shared__ int   si[256];

    float z1[4];
    float m1 = -3.4e38f, m2v = -3.4e38f;
    int   m2i = 0;
#pragma unroll
    for (int p = 0; p < 4; ++p) {
        int m = t + p * 256;
        size_t o = (size_t)n * M_CTX + m;
        float lg = logits[o];
        float g1 = -logf(-logf(u1[o] + 1e-20f) + 1e-20f);
        float g2 = -logf(-logf(u2[o] + 1e-20f) + 1e-20f);
        z1[p] = (lg + g1) * 2.0f;           // / temp(0.5)
        float z2 = (lg + g2) * 2.0f;
        m1 = fmaxf(m1, z1[p]);
        if (z2 > m2v) { m2v = z2; m2i = m; }
    }
    // row max of z1
    sv[t] = m1; __syncthreads();
    for (int s = 128; s > 0; s >>= 1) { if (t < s) sv[t] = fmaxf(sv[t], sv[t + s]); __syncthreads(); }
    float rmax = sv[0]; __syncthreads();
    // exp + sum
    float lsum = 0.f;
#pragma unroll
    for (int p = 0; p < 4; ++p) { z1[p] = __expf(z1[p] - rmax); lsum += z1[p]; }
    sv[t] = lsum; __syncthreads();
    for (int s = 128; s > 0; s >>= 1) { if (t < s) sv[t] += sv[t + s]; __syncthreads(); }
    float inv = 1.0f / sv[0]; __syncthreads();
#pragma unroll
    for (int p = 0; p < 4; ++p) {
        int m = t + p * 256;
        asgT[(size_t)m * N_SMP + n] = f2bf(z1[p] * inv);
    }
    // argmax (first index on ties, matching jnp.argmax)
    sv[t] = m2v; si[t] = m2i; __syncthreads();
    for (int s = 128; s > 0; s >>= 1) {
        if (t < s) {
            if (sv[t + s] > sv[t] || (sv[t + s] == sv[t] && si[t + s] < si[t])) {
                sv[t] = sv[t + s]; si[t] = si[t + s];
            }
        }
        __syncthreads();
    }
    int am = si[0];
#pragma unroll
    for (int p = 0; p < 4; ++p) {
        int m = t + p * 256;
        hard[(size_t)n * M_CTX + m] = (m == am) ? 1.0f : 0.0f;
    }
}

// ---------------------------------------------------------------------------
// LayerNorm in place over rows of D_MODEL (block = row)
// ---------------------------------------------------------------------------
__global__ __launch_bounds__(256)
void layernorm_kernel(float* __restrict__ x,
                      const float* __restrict__ gamma, const float* __restrict__ beta) {
    const int row = blockIdx.x;
    const int t   = threadIdx.x;
    __shared__ float s1[256], s2[256];
    float a = 0.f, b = 0.f;
#pragma unroll
    for (int p = 0; p < 4; ++p) {
        float v = x[(size_t)row * D_MODEL + t + p * 256];
        a += v; b += v * v;
    }
    s1[t] = a; s2[t] = b; __syncthreads();
    for (int s = 128; s > 0; s >>= 1) {
        if (t < s) { s1[t] += s1[t + s]; s2[t] += s2[t + s]; }
        __syncthreads();
    }
    float mu  = s1[0] * (1.0f / D_MODEL);
    float var = s2[0] * (1.0f / D_MODEL) - mu * mu;
    float inv = rsqrtf(var + 1e-3f);
#pragma unroll
    for (int p = 0; p < 4; ++p) {
        int j = t + p * 256;
        size_t o = (size_t)row * D_MODEL + j;
        x[o] = (x[o] - mu) * inv * gamma[j] + beta[j];
    }
}

// ---------------------------------------------------------------------------
// Launcher
// ---------------------------------------------------------------------------
extern "C" void kernel_launch(void* const* d_in, const int* in_sizes, int n_in,
                              void* d_out, int out_size, void* d_ws, size_t ws_size,
                              hipStream_t stream) {
    const float* context = (const float*)d_in[0];
    const float* sample  = (const float*)d_in[1];
    const float* Wq = (const float*)d_in[2];  const float* bq = (const float*)d_in[3];
    const float* Wk = (const float*)d_in[4];  const float* bk = (const float*)d_in[5];
    const float* Wv = (const float*)d_in[6];  const float* bv = (const float*)d_in[7];
    const float* gamma = (const float*)d_in[8];
    const float* beta  = (const float*)d_in[9];
    const float* u1 = (const float*)d_in[10];
    const float* u2 = (const float*)d_in[11];

    // workspace layout
    char* base = (char*)d_ws;
    size_t off = 0;
    auto alloc = [&](size_t bytes) { size_t o = off; off += (bytes + 255) & ~(size_t)255; return o; };
    const size_t oS   = alloc((size_t)N_SMP * D_IN    * 2);  // sample bf16 (reused as asgT)
    const size_t oC   = alloc((size_t)M_CTX * D_IN    * 2);  // context bf16
    const size_t oWq  = alloc((size_t)D_IN  * D_MODEL * 2);  // Wq^T bf16
    const size_t oWk  = alloc((size_t)D_IN  * D_MODEL * 2);
    const size_t oWv  = alloc((size_t)D_IN  * D_MODEL * 2);
    const size_t oQ   = alloc((size_t)M_CTX * D_MODEL * 2);  // q bf16
    const size_t oK   = alloc((size_t)N_SMP * D_MODEL * 2);  // k bf16
    const size_t oVT  = alloc((size_t)D_MODEL * N_SMP * 2);  // v^T bf16
    const size_t oLG  = alloc((size_t)N_SMP * M_CTX   * 4);  // logits f32

    u16*  Sb   = (u16*)(base + oS);
    u16*  Cb   = (u16*)(base + oC);
    u16*  Wqt  = (u16*)(base + oWq);
    u16*  Wkt  = (u16*)(base + oWk);
    u16*  Wvt  = (u16*)(base + oWv);
    u16*  Qb   = (u16*)(base + oQ);
    u16*  Kb   = (u16*)(base + oK);
    u16*  VTb  = (u16*)(base + oVT);
    float* LG  = (float*)(base + oLG);
    u16*  AsgT = Sb;                                  // alias: sample bf16 dead by then

    float* aggOut  = (float*)d_out;                               // [M_CTX, D_MODEL]
    float* hardOut = (float*)d_out + (size_t)M_CTX * D_MODEL;     // [N_SMP, M_CTX]

    // 1) casts
    cast_bf16_kernel<<<(N_SMP * D_IN) / 256, 256, 0, stream>>>(sample, Sb, N_SMP * D_IN);
    cast_bf16_kernel<<<(M_CTX * D_IN) / 256, 256, 0, stream>>>(context, Cb, M_CTX * D_IN);
    transpose_cast_kernel<<<(D_IN * D_MODEL) / 256, 256, 0, stream>>>(Wq, Wqt, D_IN, D_MODEL);
    transpose_cast_kernel<<<(D_IN * D_MODEL) / 256, 256, 0, stream>>>(Wk, Wkt, D_IN, D_MODEL);
    transpose_cast_kernel<<<(D_IN * D_MODEL) / 256, 256, 0, stream>>>(Wv, Wvt, D_IN, D_MODEL);

    // 2) projections (C = A * W^T^T + b via Bt form)
    dim3 gQ(D_MODEL / BN, M_CTX / BM);
    gemm_bt_kernel<<<gQ, 256, 0, stream>>>(Cb, D_IN, Wqt, D_IN, bq,
                                           nullptr, 0, Qb, D_MODEL, 0, D_IN);
    dim3 gKV(D_MODEL / BN, N_SMP / BM);
    gemm_bt_kernel<<<gKV, 256, 0, stream>>>(Sb, D_IN, Wkt, D_IN, bk,
                                            nullptr, 0, Kb, D_MODEL, 0, D_IN);
    gemm_bt_kernel<<<gKV, 256, 0, stream>>>(Sb, D_IN, Wvt, D_IN, bv,
                                            nullptr, 0, VTb, N_SMP, /*trans*/1, D_IN);

    // 3) logits[n][m] = sum_d k[n][d] * q[m][d]   (A=k, B=q)
    dim3 gLG(M_CTX / BN, N_SMP / BM);
    gemm_bt_kernel<<<gLG, 256, 0, stream>>>(Kb, D_MODEL, Qb, D_MODEL, nullptr,
                                            LG, M_CTX, nullptr, 0, 0, D_MODEL);

    // 4) gumbel softmax (soft -> AsgT bf16 transposed; hard -> one-hot f32)
    gumbel_kernel<<<N_SMP, 256, 0, stream>>>(LG, u1, u2, AsgT, hardOut);

    // 5) agg[m][d] = sum_n asg[n][m] * v[n][d]  (A=AsgT [M,N_SMP], B=VT [D,N_SMP])
    dim3 gAG(D_MODEL / BN, M_CTX / BM);
    gemm_bt_kernel<<<gAG, 256, 0, stream>>>(AsgT, N_SMP, VTb, N_SMP, nullptr,
                                            aggOut, D_MODEL, nullptr, 0, 0, N_SMP);

    // 6) layernorm in place on agg output
    layernorm_kernel<<<M_CTX, 256, 0, stream>>>(aggOut, gamma, beta);
}
